// PRover_85401129713827
// MI455X (gfx1250) — matrix operand order repaired and checked by reading, not compiled, not run on hardware
//
#include <hip/hip_runtime.h>
#include <cstdint>

// ---------------------------------------------------------------------------
// Shapes (fixed by the reference): B=16, S=512, H=1024, K=32, E=1024, L=2
// ---------------------------------------------------------------------------

typedef __attribute__((ext_vector_type(16))) __bf16 v16bf;
typedef __attribute__((ext_vector_type(8)))  __bf16 v8bf;
typedef __attribute__((ext_vector_type(8)))  float  v8f;

struct BPack { uint4 lo, hi; };

__device__ __forceinline__ v16bf load_frag16(const __bf16* p0, const __bf16* p1) {
    BPack pk;
    pk.lo = *reinterpret_cast<const uint4*>(p0);
    pk.hi = *reinterpret_cast<const uint4*>(p1);
    return __builtin_bit_cast(v16bf, pk);
}

__device__ __forceinline__ v8f wmma_bf16(v16bf a, v16bf b, v8f c) {
    return __builtin_amdgcn_wmma_f32_16x16x32_bf16(false, a, false, b,
                                                   (short)0, c, false, false);
}

// ---------------------------------------------------------------------------
// LDS-staged, double-buffered block GEMM (the only WMMA kernel):
//   val = (Arow_r . Wt_c) + dense_b[c],  c in [0,1024)
//   act==1 -> dest[r][c] = bf16(tanh(val));  act==0 -> dest[r][c] = bf16(val)
// Block tile 128M x 128N; 8 waves as 4(M) x 2(N); wave tile 32x64 = 8 WMMA acc.
// A/B k-slices (128x32 bf16) staged via LDS (stride 40 elems = 80 B: 16B
// aligned, conflict-free ds_load_b128 fragments), double buffered; next-slice
// global loads are issued BEFORE the wmma block so their wait lands after it.
// gather==1: edge rows r'=(rowOfs+r): b=r'/1024, e=r'%1024, m=nm[b], i=e/m,
//            j=e%m, A = [fn[b][j] | fn[b][i] | fn[b][j]-fn[b][i]],
//            zero row if e >= m*m.
// ---------------------------------------------------------------------------
__global__ void __launch_bounds__(256)
gemm_tanh(const __bf16* __restrict__ A, const __bf16* __restrict__ Wt,
          const float* __restrict__ dense_b, __bf16* __restrict__ dest,
          int Mrows, int KD, int act, int gather, int rowOfs,
          const __bf16* __restrict__ fn, const int* __restrict__ nm) {
    __shared__ __attribute__((aligned(16))) __bf16 Abuf[2][128 * 40];
    __shared__ __attribute__((aligned(16))) __bf16 Bbuf[2][128 * 40];

    const int t    = threadIdx.x;
    const int lane = t & 31;
    const int wave = t >> 5;
    const int half = lane >> 4;
    const int lq   = lane & 15;
    const int wm   = wave & 3;        // 0..3  (M subgroup, 32 rows each)
    const int wn   = wave >> 2;       // 0..1  (N subgroup, 64 cols each)
    const int rowBlock = blockIdx.x * 128;
    const int nBlock   = blockIdx.y * 128;

    // ---- staging descriptors: thread t owns (slot sr, k-half ks) ----
    const int sr = t >> 1;            // 0..127
    const int ks = (t & 1) * 16;      // 0 | 16
    const __bf16* stA = nullptr;
    const __bf16* stJ = nullptr;
    const __bf16* stI = nullptr;
    bool stValid = true;
    if (gather) {
        const int ar = rowOfs + rowBlock + sr;
        const int b = ar >> 10, e = ar & 1023;
        const int m = nm[b * 2 + 1];
        const int i = e / m;
        const int j = e - i * m;
        stValid = (e < m * m);
        stJ = fn + ((size_t)b * 32 + j) * 1024;
        stI = fn + ((size_t)b * 32 + i) * 1024;
    } else {
        int ar = rowBlock + sr;
        if (ar > Mrows - 1) ar = Mrows - 1;
        stA = A + (size_t)ar * KD;
    }
    const __bf16* stB = Wt + (size_t)(nBlock + sr) * KD;

    auto stageLoadA = [&](int k0, uint4& a0, uint4& a1) {
        if (!gather) {
            a0 = *reinterpret_cast<const uint4*>(stA + k0 + ks);
            a1 = *reinterpret_cast<const uint4*>(stA + k0 + ks + 8);
            return;
        }
        if (!stValid) { a0 = make_uint4(0,0,0,0); a1 = make_uint4(0,0,0,0); return; }
        const int sec = k0 >> 10;               // 32 | 1024: never straddles
        const int kk  = (k0 & 1023) + ks;
        if (sec == 0) {
            a0 = *reinterpret_cast<const uint4*>(stJ + kk);
            a1 = *reinterpret_cast<const uint4*>(stJ + kk + 8);
        } else if (sec == 1) {
            a0 = *reinterpret_cast<const uint4*>(stI + kk);
            a1 = *reinterpret_cast<const uint4*>(stI + kk + 8);
        } else {
            v8bf j0 = *reinterpret_cast<const v8bf*>(stJ + kk);
            v8bf j1 = *reinterpret_cast<const v8bf*>(stJ + kk + 8);
            v8bf i0 = *reinterpret_cast<const v8bf*>(stI + kk);
            v8bf i1 = *reinterpret_cast<const v8bf*>(stI + kk + 8);
            v8bf d0, d1;
            for (int q = 0; q < 8; ++q) {
                d0[q] = (__bf16)((float)j0[q] - (float)i0[q]);
                d1[q] = (__bf16)((float)j1[q] - (float)i1[q]);
            }
            a0 = __builtin_bit_cast(uint4, d0);
            a1 = __builtin_bit_cast(uint4, d1);
        }
    };
    auto stageLoadB = [&](int k0, uint4& b0, uint4& b1) {
        b0 = *reinterpret_cast<const uint4*>(stB + k0 + ks);
        b1 = *reinterpret_cast<const uint4*>(stB + k0 + ks + 8);
    };
    auto stageStore = [&](int buf, const uint4& a0, const uint4& a1,
                          const uint4& b0, const uint4& b1) {
        __bf16* pa = &Abuf[buf][sr * 40 + ks];
        *reinterpret_cast<uint4*>(pa)     = a0;
        *reinterpret_cast<uint4*>(pa + 8) = a1;
        __bf16* pb = &Bbuf[buf][sr * 40 + ks];
        *reinterpret_cast<uint4*>(pb)     = b0;
        *reinterpret_cast<uint4*>(pb + 8) = b1;
    };
    auto ldsFragA = [&](int buf, int mi) -> v16bf {
        const __bf16* p = &Abuf[buf][(wm * 32 + mi * 16 + lq) * 40];
        return load_frag16(p + 8 * half, p + 16 + 8 * half);
    };
    auto ldsFragB = [&](int buf, int ni) -> v16bf {
        const __bf16* p = &Bbuf[buf][(wn * 64 + ni * 16 + lq) * 40];
        return load_frag16(p + 8 * half, p + 16 + 8 * half);
    };

    v8f acc[8];                                  // acc[mi*4 + ni]
    for (int x = 0; x < 8; ++x)
        for (int q = 0; q < 8; ++q) acc[x][q] = 0.0f;

    {   // prologue: stage k-slice 0 into buffer 0
        uint4 a0, a1, b0, b1;
        stageLoadA(0, a0, a1);
        stageLoadB(0, b0, b1);
        stageStore(0, a0, a1, b0, b1);
    }
    __syncthreads();

    for (int k0 = 0; k0 < KD; k0 += 32) {
        const int cur = (k0 >> 5) & 1;
        const int nxt = cur ^ 1;
        const bool more = (k0 + 32) < KD;
        uint4 na0, na1, nb0, nb1;
        if (more) {                 // issue next-slice global loads first
            stageLoadA(k0 + 32, na0, na1);
            stageLoadB(k0 + 32, nb0, nb1);
        }
        // compute on current LDS buffer (overlaps the loads above)
        v16bf af0 = ldsFragA(cur, 0);
        v16bf af1 = ldsFragA(cur, 1);
        #pragma unroll
        for (int ni = 0; ni < 4; ++ni) {
            v16bf bf = ldsFragB(cur, ni);
            acc[ni]     = wmma_bf16(af0, bf, acc[ni]);
            acc[4 + ni] = wmma_bf16(af1, bf, acc[4 + ni]);
        }
        if (more) stageStore(nxt, na0, na1, nb0, nb1);
        __syncthreads();
    }

    // epilogue: dest = (tanh?)(acc + dense_b[col]) as bf16; guard r >= Mrows
    for (int mi = 0; mi < 2; ++mi) {
        const int rbase = rowBlock + wm * 32 + mi * 16 + 8 * half;
        for (int ni = 0; ni < 4; ++ni) {
            const int col = nBlock + wn * 64 + ni * 16 + lq;
            const float bv = dense_b[col];
            for (int q = 0; q < 8; ++q) {
                const int r = rbase + q;
                if (r < Mrows) {
                    float v = acc[mi * 4 + ni][q] + bv;
                    if (act) v = tanhf(v);
                    dest[(size_t)r * 1024 + col] = (__bf16)v;
                }
            }
        }
    }
}

// ---------------------------------------------------------------------------
// Tiled transpose + f32 -> bf16 convert:  src[K][N] f32  ->  dst[N][K] bf16
// ---------------------------------------------------------------------------
__global__ void __launch_bounds__(256)
transpose_bf16(const float* __restrict__ src, __bf16* __restrict__ dst,
               int K, int N) {
    __shared__ __bf16 tile[32][33];
    const int kt = blockIdx.y * 32;
    const int nt = blockIdx.x * 32;
    const int x  = threadIdx.x;
    for (int yy = threadIdx.y; yy < 32; yy += 8)
        tile[yy][x] = (__bf16)src[(size_t)(kt + yy) * N + (nt + x)];
    __syncthreads();
    for (int yy = threadIdx.y; yy < 32; yy += 8)
        dst[(size_t)(nt + yy) * K + (kt + x)] = tile[x][yy];
}

// ---------------------------------------------------------------------------
// cls_tok (16 x 1024) bf16 from seq[:,0,:]
// ---------------------------------------------------------------------------
__global__ void __launch_bounds__(256)
cvt_clstok(const float* __restrict__ seq, __bf16* __restrict__ dst) {
    const int idx = blockIdx.x * 256 + threadIdx.x;   // 16384
    const int b = idx >> 10, h = idx & 1023;
    dst[idx] = (__bf16)seq[(size_t)b * 512 * 1024 + h];
}

// ---------------------------------------------------------------------------
// n[b], m[b] = n+1 from proof_offsets (int32, 16 x 32)
// ---------------------------------------------------------------------------
__global__ void compute_nm(const int* __restrict__ po, int* __restrict__ nm) {
    const int b = threadIdx.x;
    if (b < 16) {
        int n = 0;
        for (int k = 1; k < 32; ++k) n += (po[b * 32 + k] > 0) ? 1 : 0;
        nm[b * 2 + 0] = n;
        nm[b * 2 + 1] = n + 1;
    }
}

// ---------------------------------------------------------------------------
// full_nodes bf16 (16 x 32 x 1024): segment means / naf row / zeros
// ---------------------------------------------------------------------------
__global__ void __launch_bounds__(256)
build_fn(const float* __restrict__ seq, const int* __restrict__ po,
         const int* __restrict__ nm, const __bf16* __restrict__ nafb,
         __bf16* __restrict__ fn) {
    const int blk = blockIdx.x;            // 512 = 16 * 32
    const int b = blk >> 5, k = blk & 31;
    const int n = nm[b * 2];
    __bf16* dst = fn + (size_t)(b * 32 + k) * 1024;
    const int t = threadIdx.x;

    if (k > n) {
        for (int h = t; h < 1024; h += 256) dst[h] = (__bf16)0.0f;
        return;
    }
    if (k == n) {
        const __bf16* s = nafb + (size_t)b * 1024;
        for (int h = t; h < 1024; h += 256) dst[h] = s[h];
        return;
    }
    int end   = po[b * 32 + k + 1];
    int start = (k == 0) ? 1 : (po[b * 32 + k] + 1);
    if (start < 0)  start = 0;
    if (end > 511)  end = 511;
    const int cnt = end - start + 1;
    const float inv = 1.0f / (float)(cnt > 1 ? cnt : 1);
    const float* sb = seq + (size_t)b * 512 * 1024;
    for (int h = t; h < 1024; h += 256) {
        float s = 0.0f;
        for (int r = start; r <= end; ++r) s += sb[(size_t)r * 1024 + h];
        dst[h] = (__bf16)(s * inv);
    }
}

// ---------------------------------------------------------------------------
// Skinny out-projection: out[r][l] = hid[r][:] . outW[:][l] + out_b[l], l=0,1
// One wave per row; lane covers 4x 8-elem chunks; full-wave shfl reduction.
// ---------------------------------------------------------------------------
__global__ void __launch_bounds__(256)
proj_head(const __bf16* __restrict__ hid, int Mrows,
          const float* __restrict__ oW, const float* __restrict__ ob,
          float* __restrict__ out) {
    const int lane = threadIdx.x & 31;
    const int wave = threadIdx.x >> 5;
    const int row  = blockIdx.x * 8 + wave;
    if (row >= Mrows) return;

    const __bf16* hp = hid + (size_t)row * 1024;
    float s0 = 0.0f, s1 = 0.0f;
    for (int c = 0; c < 4; ++c) {
        const int h0 = c * 256 + lane * 8;
        v8bf v = *reinterpret_cast<const v8bf*>(hp + h0);
        for (int q = 0; q < 8; ++q) {
            const float f = (float)v[q];
            s0 += f * oW[(h0 + q) * 2 + 0];
            s1 += f * oW[(h0 + q) * 2 + 1];
        }
    }
    for (int off = 1; off < 32; off <<= 1) {
        s0 += __shfl_xor(s0, off, 32);
        s1 += __shfl_xor(s1, off, 32);
    }
    if (lane == 0) {
        float2 r;
        r.x = s0 + ob[0];
        r.y = s1 + ob[1];
        *reinterpret_cast<float2*>(out + (size_t)row * 2) = r;
    }
}

// ---------------------------------------------------------------------------
// Host launcher
// ---------------------------------------------------------------------------
extern "C" void kernel_launch(void* const* d_in, const int* in_sizes, int n_in,
                              void* d_out, int out_size, void* d_ws, size_t ws_size,
                              hipStream_t stream) {
    (void)in_sizes; (void)n_in; (void)out_size; (void)ws_size;

    const float* seq     = (const float*)d_in[0];
    const int*   po      = (const int*)  d_in[1];
    const float* naf_W   = (const float*)d_in[2];
    const float* naf_b   = (const float*)d_in[3];
    const float* cls_dW  = (const float*)d_in[4];
    const float* cls_db  = (const float*)d_in[5];
    const float* cls_oW  = (const float*)d_in[6];
    const float* cls_ob  = (const float*)d_in[7];
    const float* node_dW = (const float*)d_in[8];
    const float* node_db = (const float*)d_in[9];
    const float* node_oW = (const float*)d_in[10];
    const float* node_ob = (const float*)d_in[11];
    const float* edge_dW = (const float*)d_in[12];
    const float* edge_db = (const float*)d_in[13];
    const float* edge_oW = (const float*)d_in[14];
    const float* edge_ob = (const float*)d_in[15];

    char* ws = (char*)d_ws;
    const size_t MB = 1u << 20;
    __bf16* nafWt  = (__bf16*)(ws + 0 * MB);          // 1024x1024 bf16 (2 MB)
    __bf16* clsWt  = (__bf16*)(ws + 2 * MB);          // 2 MB
    __bf16* nodeWt = (__bf16*)(ws + 4 * MB);          // 2 MB
    __bf16* edgeWt = (__bf16*)(ws + 6 * MB);          // 1024x3072 bf16 (6 MB)
    __bf16* clstok = (__bf16*)(ws + 12 * MB);         // 16x1024 (32 KB)
    __bf16* nafv   = (__bf16*)(ws + 12 * MB + 65536); // 16x1024 (32 KB)
    __bf16* fn     = (__bf16*)(ws + 13 * MB);         // 16x32x1024 (1 MB)
    int*    nm     = (int*)   (ws + 14 * MB);
    __bf16* hid    = (__bf16*)(ws + 15 * MB);         // 8192x1024 bf16 (16 MB)

    const dim3 tb(32, 8);
    // Weight transpose + bf16 convert (one-time per call; L2-resident after)
    transpose_bf16<<<dim3(32, 32), tb, 0, stream>>>(naf_W,   nafWt,  1024, 1024);
    transpose_bf16<<<dim3(32, 32), tb, 0, stream>>>(cls_dW,  clsWt,  1024, 1024);
    transpose_bf16<<<dim3(32, 32), tb, 0, stream>>>(node_dW, nodeWt, 1024, 1024);
    transpose_bf16<<<dim3(32, 96), tb, 0, stream>>>(edge_dW, edgeWt, 3072, 1024);

    cvt_clstok<<<64, 256, 0, stream>>>(seq, clstok);
    compute_nm<<<1, 32, 0, stream>>>(po, nm);

    // naf = cls_tok @ naf_W + naf_b  (linear, bf16 out) via the same GEMM
    gemm_tanh<<<dim3(1, 8), 256, 0, stream>>>(clstok, nafWt, naf_b, nafv,
                                              16, 1024, /*act=*/0, 0, 0,
                                              nullptr, nullptr);
    build_fn<<<512, 256, 0, stream>>>(seq, po, nm, nafv, fn);

    float* out = (float*)d_out;
    // cls head: M=16
    gemm_tanh<<<dim3(1, 8), 256, 0, stream>>>(clstok, clsWt, cls_db, hid,
                                              16, 1024, 1, 0, 0, nullptr, nullptr);
    proj_head<<<2, 256, 0, stream>>>(hid, 16, cls_oW, cls_ob, out);
    // node head: M=512
    gemm_tanh<<<dim3(4, 8), 256, 0, stream>>>(fn, nodeWt, node_db, hid,
                                              512, 1024, 1, 0, 0, nullptr, nullptr);
    proj_head<<<64, 256, 0, stream>>>(hid, 512, node_oW, node_ob, out + 32);
    // edge head: M=16384 in two 8192-row chunks (KD=3072, gathered A)
    for (int c = 0; c < 2; ++c) {
        gemm_tanh<<<dim3(64, 8), 256, 0, stream>>>(nullptr, edgeWt, edge_db, hid,
                                                   8192, 3072, 1, 1, c * 8192,
                                                   fn, nm);
        proj_head<<<1024, 256, 0, stream>>>(hid, 8192, edge_oW, edge_ob,
                                            out + 32 + 1024 + (size_t)c * 8192 * 2);
    }
}
